// CN_69312182223156
// MI455X (gfx1250) — matrix operand order repaired and checked by reading, not compiled, not run on hardware
//
#include <hip/hip_runtime.h>

typedef __attribute__((ext_vector_type(16))) _Float16 v16h;
typedef __attribute__((ext_vector_type(8)))  float    v8f;
typedef _Float16 f16;

#define DIM   32
#define LS    100
#define NTILE 7            // ceil(100/16), l padded to 112
#define HID   50
#define NP    9696         // 3*32*101
#define DL    3200         // 32*100
#define BATCH 16384

// packed operand sizes (in halves)
#define PW2_HALVES (32 * NTILE * 3 * 2 * 512)   // [d][t][reg][ks][lane][16]
#define PW1_HALVES (4 * 512)                    // [t][lane][16]
#define PTL_HALVES (32 * 2 * 512)               // [d][ks][lane][16]

// ---------------- prep kernels: f32 -> f16, WMMA-B-ready layout ----------------
// B operand k order used here, same as A (16x16x32 f16, 32x16 B): lane half hi covers K block hi*16..hi*16+15,
// VGPR v holds K = hi*16 + 2v + h.

__global__ void pack_w2(const float* __restrict__ W2, f16* __restrict__ outp) {
    int idx = blockIdx.x * blockDim.x + threadIdx.x;
    if (idx >= PW2_HALVES) return;
    int j    = idx & 15;
    int lane = (idx >> 4) & 31;
    int ks   = (idx >> 9) & 1;
    int r3   = idx >> 10;
    int reg  = r3 % 3;
    int r4   = r3 / 3;
    int t    = r4 % NTILE;
    int d    = r4 / NTILE;
    int n = lane & 15, hi = lane >> 4, v = j >> 1, hh = j & 1;
    int kk = ks * 32 + hi * 16 + v * 2 + hh;
    int l  = t * 16 + n;
    float val = 0.f;
    if (kk < HID && l < LS) {
        int c = reg * DL + d * LS + l;
        val = W2[kk * NP + c];
    }
    outp[idx] = (f16)val;
}

__global__ void pack_w1(const float* __restrict__ W1, f16* __restrict__ outp) {
    int idx = blockIdx.x * blockDim.x + threadIdx.x;
    if (idx >= PW1_HALVES) return;
    int j    = idx & 15;
    int lane = (idx >> 4) & 31;
    int t    = idx >> 9;            // 0..3
    int n = lane & 15, hi = lane >> 4, v = j >> 1, hh = j & 1;
    int kk  = hi * 16 + v * 2 + hh; // K = 32 exact
    int col = t * 16 + n;
    float val = (col < HID) ? W1[kk * HID + col] : 0.f;
    outp[idx] = (f16)val;
}

__global__ void pack_tail(const float* __restrict__ W2, f16* __restrict__ outp) {
    int idx = blockIdx.x * blockDim.x + threadIdx.x;
    if (idx >= PTL_HALVES) return;
    int j    = idx & 15;
    int lane = (idx >> 4) & 31;
    int ks   = (idx >> 9) & 1;
    int d    = idx >> 10;
    int n = lane & 15, hi = lane >> 4, v = j >> 1, hh = j & 1;
    int kk = ks * 32 + hi * 16 + v * 2 + hh;
    float val = 0.f;
    if (kk < HID && n < 3) {
        int cg = 3 * DL + n * 32 + d;   // n=0: bias2, n=1: eps, n=2: alpha
        val = W2[kk * NP + cg];
    }
    outp[idx] = (f16)val;
}

// ---------------- fused main kernel ----------------
static __device__ inline v8f wmma_acc(v16h a, v16h b, v8f c) {
    return __builtin_amdgcn_wmma_f32_16x16x32_f16(false, a, false, b, (short)0, c, false, false);
}

__global__ __launch_bounds__(1024, 1)
void pfnd_main(const float* __restrict__ x,
               const float* __restrict__ b1_0, const float* __restrict__ b2_0,
               const float* __restrict__ b1_1, const float* __restrict__ b2_1,
               const f16* __restrict__ pw2_0, const f16* __restrict__ pw2_1,
               const f16* __restrict__ pw1_0, const f16* __restrict__ pw1_1,
               const f16* __restrict__ ptl_0, const f16* __restrict__ ptl_1,
               float* __restrict__ out)
{
    __shared__ float xt[16][64];   // input row tile [16 rows][x1|x2]
    __shared__ float ht[16][64];   // hidden h, K padded to 64 (zeros beyond 50)
    __shared__ float y1t[16][32];
    __shared__ float y2t[16][32];

    const int tid  = threadIdx.x;
    const int lane = tid & 31;
    const int wave = tid >> 5;       // wave index == feature dim d
    const int row0 = blockIdx.x * 16;

    { int r = tid >> 6, c = tid & 63;
      xt[r][c] = x[(row0 + r) * 64 + c]; }
    __syncthreads();

    const int n  = lane & 15;
    const int hi = lane >> 4;
    const int hibase = hi * 8;       // C-layout row base for this lane half
    const int d  = wave;

    const float* b1s[2] = { b1_0, b1_1 };
    const float* b2s[2] = { b2_0, b2_1 };
    const f16* pw2s[2]  = { pw2_0, pw2_1 };
    const f16* pw1s[2]  = { pw1_0, pw1_1 };
    const f16* ptls[2]  = { ptl_0, ptl_1 };

    for (int s = 0; s < 2; ++s) {
        const float* cond = (s == 0) ? &xt[0][32] : &y1t[0][0];
        const int cstride = (s == 0) ? 64 : 32;
        const float* xdat = (s == 0) ? &xt[0][0] : &xt[0][32];   // stride 64
        float (*yt)[32]   = (s == 0) ? y1t : y2t;
        const float* b1 = b1s[s];
        const float* b2 = b2s[s];

        // ---- Step A: h = relu(cond @ W1 + b1), waves 0..3 each one 16-col tile ----
        if (wave < 4) {
            v16h aC;
            #pragma unroll
            for (int j = 0; j < 16; ++j) {
                int v = j >> 1, hh = j & 1;
                int kk = hi * 16 + v * 2 + hh;  // A layout
                aC[j] = (f16)cond[n * cstride + kk];
            }
            v16h bW = *(const v16h*)(pw1s[s] + wave * 512 + lane * 16);
            v8f c = {};
            c = wmma_acc(aC, bW, c);
            int col = wave * 16 + n;
            float b1v = (col < HID) ? b1[col] : 0.f;
            #pragma unroll
            for (int r = 0; r < 8; ++r) {
                float hv = (col < HID) ? fmaxf(c[r] + b1v, 0.f) : 0.f;
                ht[hibase + r][col] = hv;
            }
        }
        __syncthreads();

        // ---- Step B: A operand (h, M=16, K=64) from LDS ----
        v16h a0, a1;
        #pragma unroll
        for (int j = 0; j < 16; ++j) {
            int v = j >> 1, hh = j & 1;
            int kk = hi * 16 + v * 2 + hh;
            a0[j] = (f16)ht[n][kk];
            a1[j] = (f16)ht[n][32 + kk];
        }

        float xd[8];
        #pragma unroll
        for (int r = 0; r < 8; ++r) xd[r] = xdat[(hibase + r) * 64 + d];

        float accN[8], accD[8];
        #pragma unroll
        for (int r = 0; r < 8; ++r) { accN[r] = 0.f; accD[r] = 0.f; }

        const f16* wb = pw2s[s] + (size_t)d * (NTILE * 3 * 2 * 512);
        for (int t = 0; t < NTILE; ++t) {
            const f16* tb = wb + t * (3 * 2 * 512);
            v16h bm10 = *(const v16h*)(tb + 0 * 1024 +   0 + lane * 16);
            v16h bm11 = *(const v16h*)(tb + 0 * 1024 + 512 + lane * 16);
            v16h bb10 = *(const v16h*)(tb + 1 * 1024 +   0 + lane * 16);
            v16h bb11 = *(const v16h*)(tb + 1 * 1024 + 512 + lane * 16);
            v16h bm20 = *(const v16h*)(tb + 2 * 1024 +   0 + lane * 16);
            v16h bm21 = *(const v16h*)(tb + 2 * 1024 + 512 + lane * 16);
            v8f cm1 = {}, cb1 = {}, cm2 = {};
            cm1 = wmma_acc(a0, bm10, cm1); cm1 = wmma_acc(a1, bm11, cm1);
            cb1 = wmma_acc(a0, bb10, cb1); cb1 = wmma_acc(a1, bb11, cb1);
            cm2 = wmma_acc(a0, bm20, cm2); cm2 = wmma_acc(a1, bm21, cm2);

            int l = t * 16 + n;
            if (l < LS) {
                int c = d * LS + l;
                float bm1v = b2[c];
                float bbiv = b2[DL + c];
                float bm2v = b2[2 * DL + c];
                #pragma unroll
                for (int r = 0; r < 8; ++r) {
                    float m1 = cm1[r] + bm1v;
                    float bi = cb1[r] + bbiv;
                    float m2 = cm2[r] + bm2v;
                    float z  = fmaf(xd[r], m1, bi);
                    float e  = (z > 0.f) ? z : (__expf(z) - 1.f);   // elu
                    accN[r] += e * m2;
                    accD[r] += fmaxf(-m1 * m2, 0.f);
                }
            }
        }

        // ---- tail tile: bias2 (n=0), eps (n=1), alpha (n=2) columns ----
        float tv[8];
        {
            const f16* tb = ptls[s] + d * 1024;
            v16h t0 = *(const v16h*)(tb + lane * 16);
            v16h t1 = *(const v16h*)(tb + 512 + lane * 16);
            v8f ct = {};
            ct = wmma_acc(a0, t0, ct);
            ct = wmma_acc(a1, t1, ct);
            float btail = (n < 3) ? b2[3 * DL + n * 32 + d] : 0.f;
            #pragma unroll
            for (int r = 0; r < 8; ++r) tv[r] = ct[r] + btail;
        }

        // ---- reduce num/den over l (16 lanes per half hold disjoint l's) ----
        #pragma unroll
        for (int off = 8; off >= 1; off >>= 1) {
            #pragma unroll
            for (int r = 0; r < 8; ++r) {
                accN[r] += __shfl_xor(accN[r], off, 32);
                accD[r] += __shfl_xor(accD[r], off, 32);
            }
        }

        const int base = hi << 4;
        #pragma unroll
        for (int r = 0; r < 8; ++r) {
            float b2v = __shfl(tv[r], base + 0, 32);
            float ep  = __shfl(tv[r], base + 1, 32) * 0.1f;
            float al  = __shfl(tv[r], base + 2, 32) * 0.1f;
            if (n == 0) {
                float den = accD[r] + 1.f;
                float sig = 1.f / (1.f + __expf(-ep));
                float yv  = __expf(al) * (xd[r] + 0.8f * sig * accN[r] / den) + b2v;
                yt[hibase + r][d] = yv;
            }
        }
        __syncthreads();
    }

    { int r = tid >> 6, c = tid & 63;
      float val = (c < 32) ? y1t[r][c] : y2t[r][c - 32];
      out[(row0 + r) * 64 + c] = val; }
}

// ---------------- launch ----------------
extern "C" void kernel_launch(void* const* d_in, const int* in_sizes, int n_in,
                              void* d_out, int out_size, void* d_ws, size_t ws_size,
                              hipStream_t stream) {
    const float* x    = (const float*)d_in[0];
    const float* W1_0 = (const float*)d_in[1];
    const float* b1_0 = (const float*)d_in[2];
    const float* W2_0 = (const float*)d_in[3];
    const float* b2_0 = (const float*)d_in[4];
    const float* W1_1 = (const float*)d_in[5];
    const float* b1_1 = (const float*)d_in[6];
    const float* W2_1 = (const float*)d_in[7];
    const float* b2_1 = (const float*)d_in[8];
    float* out = (float*)d_out;

    char* ws = (char*)d_ws;
    const size_t pw2_b = (size_t)PW2_HALVES * sizeof(f16);
    const size_t pw1_b = (size_t)PW1_HALVES * sizeof(f16);
    const size_t ptl_b = (size_t)PTL_HALVES * sizeof(f16);

    f16* pw2_0 = (f16*)(ws);
    f16* pw2_1 = (f16*)(ws + pw2_b);
    f16* pw1_0 = (f16*)(ws + 2 * pw2_b);
    f16* pw1_1 = (f16*)(ws + 2 * pw2_b + pw1_b);
    f16* ptl_0 = (f16*)(ws + 2 * pw2_b + 2 * pw1_b);
    f16* ptl_1 = (f16*)(ws + 2 * pw2_b + 2 * pw1_b + ptl_b);

    pack_w2<<<PW2_HALVES / 256, 256, 0, stream>>>(W2_0, pw2_0);
    pack_w2<<<PW2_HALVES / 256, 256, 0, stream>>>(W2_1, pw2_1);
    pack_w1<<<PW1_HALVES / 256, 256, 0, stream>>>(W1_0, pw1_0);
    pack_w1<<<PW1_HALVES / 256, 256, 0, stream>>>(W1_1, pw1_1);
    pack_tail<<<PTL_HALVES / 256, 256, 0, stream>>>(W2_0, ptl_0);
    pack_tail<<<PTL_HALVES / 256, 256, 0, stream>>>(W2_1, ptl_1);

    pfnd_main<<<BATCH / 16, 1024, 0, stream>>>(
        x, b1_0, b2_0, b1_1, b2_1,
        pw2_0, pw2_1, pw1_0, pw1_1, ptl_0, ptl_1, out);
}
